// DistributedMoE_48576080118271
// MI455X (gfx1250) — compile-verified, matrix operand order checked by reading, not run on hardware
//
#include <hip/hip_runtime.h>
#include <math.h>

typedef __attribute__((ext_vector_type(16))) __bf16 v16bf;
typedef __attribute__((ext_vector_type(8)))  float  v8f;
typedef unsigned int v4u __attribute__((ext_vector_type(4)));
typedef int          v8i __attribute__((ext_vector_type(8)));
typedef int          v4i __attribute__((ext_vector_type(4)));

namespace {
constexpr int Ntok = 2048;   // B*T
constexpr int Dm   = 768;    // model dim
constexpr int Nexp = 8;      // experts
constexpr int Fdim = 3072;   // ffn dim
constexpr int MT   = 32;     // token rows per block tile
constexpr int FC   = 128;    // F chunk width
constexpr int KC   = 32;     // WMMA K step (bf16)

constexpr int NKC1   = Dm / KC;     // 24 K-chunks for GEMM1
constexpr int NFC    = Fdim / FC;   // 24 F-chunks
constexpr int NKC2   = Fdim / KC;   // 96 K-chunks for GEMM2
constexpr int CH1_EL = KC * FC;     // 4096 bf16 per GEMM1 chunk (8 KB)
constexpr int CH2_EL = KC * Dm;     // 24576 bf16 per GEMM2 chunk (48 KB)

// LDS pool byte offsets
constexpr unsigned OFF_ABF   = 0;                        // MT*Dm*2   = 49152
constexpr unsigned OFF_HBF   = 49152;                    // MT*FC*2   =  8192
constexpr unsigned OFF_STAGE = 57344;                    // CH2 bytes = 49152
constexpr unsigned OFF_RTOK  = 106496;                   // MT*4
constexpr unsigned OFF_RW    = 106624;                   // MT*4
constexpr unsigned POOL_SZ   = 106752;

// ws byte offsets
constexpr size_t WS_COUNTS = 0;
constexpr size_t WS_TIDX   = 256;
constexpr size_t WS_TW     = WS_TIDX + (size_t)Nexp * Ntok * 4;
constexpr size_t WS_XBF    = 131328;                               // 16B aligned
constexpr size_t WS_W1S    = WS_XBF + (size_t)Ntok * Dm * 2;       // 3.27 MB
constexpr size_t WS_W2S    = WS_W1S + (size_t)Nexp * Dm * Fdim * 2;
}

// round-to-nearest-even float -> bf16 bits
__device__ __forceinline__ unsigned short f2bf(float f) {
  unsigned u = __builtin_bit_cast(unsigned, f);
  unsigned r = u + 0x7FFFu + ((u >> 16) & 1u);
  return (unsigned short)(r >> 16);
}

// Fragment-native element coordinates (ISA 16-bit A/B layout):
//   lane = n + 16*(k&8)/8 ; elem j<8 -> K = (k&8)+j ; j>=8 -> K = 16+(k&8)+(j-8)
__device__ __forceinline__ int swz_k(int lane, int j) {
  int kh = (lane >> 4) * 8;
  return (j < 8) ? (kh + j) : (16 + kh + (j - 8));
}

// A fragment from row-major bf16 LDS (stride in elems): two 16B loads.
__device__ __forceinline__ v16bf frag_A16(const unsigned short* A, int row,
                                          int stride, int kbase, int lane) {
  const int half = (lane >> 4) * 8;
  const unsigned short* p = A + row * stride + kbase + half;
  union { v16bf v; uint4 q[2]; } r;
  r.q[0] = *(const uint4*)p;
  r.q[1] = *(const uint4*)(p + 16);
  return r.v;
}

// B fragment from pre-swizzled tile ([32 lanes][16 elems] contiguous).
__device__ __forceinline__ v16bf frag_ldB(const unsigned short* tileBase, int lane) {
  const unsigned short* p = tileBase + lane * 16;
  union { v16bf v; uint4 q[2]; } r;
  r.q[0] = *(const uint4*)p;
  r.q[1] = *(const uint4*)(p + 8);
  return r.v;
}

// Per-lane async global->LDS copy of 16 bytes (ASYNCcnt tracked).
__device__ __forceinline__ void async_copy_b128(unsigned lds_addr, const void* g) {
  unsigned long long ga = (unsigned long long)(size_t)g;
  asm volatile("global_load_async_to_lds_b128 %0, %1, off"
               :: "v"(lds_addr), "v"(ga) : "memory");
}
__device__ __forceinline__ void wait_async0() {
  asm volatile("s_wait_asynccnt 0" ::: "memory");
}

// TDM: contiguous 1-D copy of n8*8 bytes global->LDS (TENSORcnt tracked).
__device__ __forceinline__ void tdm_load_1d(unsigned lds_addr, const void* gptr,
                                            unsigned n8) {
  unsigned long long ga = (unsigned long long)(size_t)gptr;
  v4u g0;
  g0[0] = 1u;                                   // count=1, user mode
  g0[1] = lds_addr;                             // LDS byte address
  g0[2] = (unsigned)(ga & 0xFFFFFFFFu);         // global_addr[31:0]
  g0[3] = (unsigned)((ga >> 32) & 0x01FFFFFFu)  // global_addr[56:32]
          | (2u << 30);                         // type=2 (image/TDM)
  v8i g1;
  g1[0] = (int)(3u << 16);                      // mask=0, data_size=8B
  g1[1] = (int)((n8 & 0xFFFFu) << 16);          // tensor_dim0[15:0]
  g1[2] = (int)(((n8 >> 16) & 0xFFFFu) | (1u << 16)); // dim0 hi, tensor_dim1=1
  g1[3] = (int)((n8 & 0xFFFFu) << 16);          // tile_dim0 = n8 (<=6144)
  g1[4] = 1;                                    // tile_dim1=1, tile_dim2=0
  g1[5] = (int)n8;                              // tensor_dim0_stride
  g1[6] = 0;
  g1[7] = 0;
  v4i z4 = {0, 0, 0, 0};
  v8i z8 = {0, 0, 0, 0, 0, 0, 0, 0};
  __builtin_amdgcn_tensor_load_to_lds(g0, g1, z4, z4, z8, 0);
}

__global__ void moe_init_kernel(float* __restrict__ out, int total,
                                int* __restrict__ counts) {
  int i = blockIdx.x * blockDim.x + threadIdx.x;
  if (i < total) out[i] = 0.0f;
  if (i < Nexp) counts[i] = 0;
}

// Pre-pass: x -> bf16; W1/W2 -> bf16 pre-swizzled into WMMA B-fragment chunk
// order so the expert kernel can DMA chunks contiguously with the TDM.
__global__ void moe_prep_kernel(const float* __restrict__ x,
                                const float* __restrict__ W1,
                                const float* __restrict__ W2,
                                unsigned short* __restrict__ Xbf,
                                unsigned short* __restrict__ W1s,
                                unsigned short* __restrict__ W2s) {
  const int NW1 = Nexp * Dm * Fdim;           // 18.87M
  const int NW2 = Nexp * Fdim * Dm;           // 18.87M
  const int NX  = Ntok * Dm;                  // 1.57M
  int idx = blockIdx.x * blockDim.x + threadIdx.x;
  if (idx < NW1) {
    int cidx = idx / CH1_EL, r = idx - cidx * CH1_EL;
    int kc = cidx % NKC1, fc = (cidx / NKC1) % NFC, e = cidx / (NKC1 * NFC);
    int t = r >> 9, lane = (r >> 4) & 31, j = r & 15;
    int n = lane & 15;
    int k = swz_k(lane, j);
    int d = kc * KC + k, f = fc * FC + t * 16 + n;
    W1s[idx] = f2bf(W1[((size_t)e * Dm + d) * Fdim + f]);
    return;
  }
  idx -= NW1;
  if (idx < NW2) {
    int cidx = idx / CH2_EL, r = idx - cidx * CH2_EL;
    int kc = cidx % NKC2, e = cidx / NKC2;
    int t = r >> 9, lane = (r >> 4) & 31, j = r & 15;
    int n = lane & 15;
    int k = swz_k(lane, j);
    int f = kc * KC + k, d = t * 16 + n;
    W2s[idx] = f2bf(W2[((size_t)e * Fdim + f) * Dm + d]);
    return;
  }
  idx -= NW2;
  if (idx < NX) Xbf[idx] = f2bf(x[idx]);
}

// One wave per token: 8 logits, softmax, top-2, renormalize, append to lists.
__global__ void moe_router_kernel(const float* __restrict__ x,
                                  const float* __restrict__ Wr,
                                  int* __restrict__ counts,
                                  int* __restrict__ tokIdx,
                                  float* __restrict__ tokW) {
  const int lane = threadIdx.x & 31;
  const int wave = threadIdx.x >> 5;
  const int tok  = blockIdx.x * 8 + wave;
  if (tok >= Ntok) return;

  float acc[Nexp];
#pragma unroll
  for (int e = 0; e < Nexp; ++e) acc[e] = 0.0f;
  const float* xr = x + (size_t)tok * Dm;
  for (int d = lane; d < Dm; d += 32) {
    float xv = xr[d];
#pragma unroll
    for (int e = 0; e < Nexp; ++e) acc[e] += xv * Wr[e * Dm + d];
  }
#pragma unroll
  for (int e = 0; e < Nexp; ++e) {
#pragma unroll
    for (int off = 16; off > 0; off >>= 1)
      acc[e] += __shfl_xor(acc[e], off, 32);
  }
  if (lane == 0) {
    float mx = acc[0];
#pragma unroll
    for (int e = 1; e < Nexp; ++e) mx = fmaxf(mx, acc[e]);
    float p[Nexp];
#pragma unroll
    for (int e = 0; e < Nexp; ++e) p[e] = __expf(acc[e] - mx);
    int i0 = 0;
#pragma unroll
    for (int e = 1; e < Nexp; ++e) if (p[e] > p[i0]) i0 = e;
    int i1 = (i0 == 0) ? 1 : 0;
#pragma unroll
    for (int e = 0; e < Nexp; ++e) if (e != i0 && p[e] > p[i1]) i1 = e;
    float w0 = p[i0], w1 = p[i1], t = w0 + w1;
    w0 /= t; w1 /= t;
    int s0 = atomicAdd(&counts[i0], 1);
    tokIdx[i0 * Ntok + s0] = tok; tokW[i0 * Ntok + s0] = w0;
    int s1 = atomicAdd(&counts[i1], 1);
    tokIdx[i1 * Ntok + s1] = tok; tokW[i1 * Ntok + s1] = w1;
  }
}

// Grouped per-expert FFN: out[tok,:] += w * (gelu(x@W1+b1) @ W2 + b2)
__global__ __launch_bounds__(256, 1)
void moe_expert_kernel(const unsigned short* __restrict__ Xbf,
                       const unsigned short* __restrict__ W1s,
                       const unsigned short* __restrict__ W2s,
                       const float* __restrict__ b1,
                       const float* __restrict__ b2,
                       const int* __restrict__ counts,
                       const int* __restrict__ tokIdx,
                       const float* __restrict__ tokW,
                       float* __restrict__ out) {
  const int e        = blockIdx.y;
  const int tileBase = blockIdx.x * MT;
  const int cnt      = counts[e];
  if (tileBase >= cnt) return;   // uniform per block

  __shared__ __attribute__((aligned(64))) unsigned char Pool[POOL_SZ];
  unsigned short* Abf   = (unsigned short*)(Pool + OFF_ABF);
  unsigned short* Hbf   = (unsigned short*)(Pool + OFF_HBF);
  unsigned short* Stage = (unsigned short*)(Pool + OFF_STAGE);
  int*            rowTok = (int*)(Pool + OFF_RTOK);
  float*          rowW   = (float*)(Pool + OFF_RW);

  const int tid  = threadIdx.x;
  const int lane = tid & 31;
  const int wave = tid >> 5;

  if (tid < MT) {
    int slot = tileBase + tid;
    if (slot < cnt) {
      rowTok[tid] = tokIdx[e * Ntok + slot];
      rowW[tid]   = tokW[e * Ntok + slot];
    } else {
      rowTok[tid] = 0;     // valid data, weight 0 -> contributes nothing
      rowW[tid]   = 0.0f;
    }
  }
  __syncthreads();

  // Gather the 32 bf16 token rows with per-lane async copies (16B each).
  {
    const unsigned abf_base = (unsigned)(size_t)(Pool + OFF_ABF);
    for (int i = tid; i < MT * (Dm / 8); i += 256) {     // 3072 chunks
      int r = i / (Dm / 8), c8 = i - r * (Dm / 8);
      async_copy_b128(abf_base + (unsigned)i * 16u,
                      Xbf + (size_t)rowTok[r] * Dm + c8 * 8);
    }
    wait_async0();
  }
  __syncthreads();

  const unsigned stage_lds = (unsigned)(size_t)(Pool + OFF_STAGE);
  const unsigned short* W1e = W1s + (size_t)e * (NFC * NKC1) * CH1_EL;
  const unsigned short* W2e = W2s + (size_t)e * NKC2 * CH2_EL;

  v8f zf = {};
  v8f accO[2][6];
#pragma unroll
  for (int mt = 0; mt < 2; ++mt)
#pragma unroll
    for (int t = 0; t < 6; ++t) accO[mt][t] = zf;

  for (int fc = 0; fc < NFC; ++fc) {
    // ---------- GEMM1: h[32,128] = X[32,768] @ W1[:,chunk] ----------
    v8f acc1[2] = { zf, zf };
    for (int kc = 0; kc < NKC1; ++kc) {
      __syncthreads();                    // stage free of readers
      if (wave == 0) {
        tdm_load_1d(stage_lds, W1e + (size_t)(fc * NKC1 + kc) * CH1_EL,
                    (unsigned)(CH1_EL * 2 / 8));
        __builtin_amdgcn_s_wait_tensorcnt(0);
      }
      __syncthreads();                    // DMA data visible
      v16bf bfrag = frag_ldB(Stage + wave * 512, lane);
#pragma unroll
      for (int mt = 0; mt < 2; ++mt) {
        v16bf afrag = frag_A16(Abf, mt * 16 + (lane & 15), Dm, kc * KC, lane);
        acc1[mt] = __builtin_amdgcn_wmma_f32_16x16x32_bf16(
            false, afrag, false, bfrag, (short)0, acc1[mt], false, false);
      }
    }
    // bias + exact erf-GeLU, re-quantize to bf16 for GEMM2 A-matrix
    {
      const int half = (lane >> 4) * 8;
      const int col  = wave * 16 + (lane & 15);
      const float bb = b1[e * Fdim + fc * FC + col];
#pragma unroll
      for (int mt = 0; mt < 2; ++mt)
#pragma unroll
        for (int r = 0; r < 8; ++r) {
          int row = mt * 16 + half + r;
          float h = acc1[mt][r] + bb;
          float g = 0.5f * h * (1.0f + erff(h * 0.70710678118654752f));
          Hbf[row * FC + col] = f2bf(g);
        }
    }
    __syncthreads();
    // ---------- GEMM2: accO += H[32,128] @ W2[chunk,:] ----------
    for (int kk = 0; kk < FC / KC; ++kk) {
      if (wave == 0) {
        tdm_load_1d(stage_lds,
                    W2e + (size_t)(fc * (FC / KC) + kk) * CH2_EL,
                    (unsigned)(CH2_EL * 2 / 8));
        __builtin_amdgcn_s_wait_tensorcnt(0);
      }
      __syncthreads();                    // DMA data visible
#pragma unroll
      for (int mt = 0; mt < 2; ++mt) {
        v16bf afrag = frag_A16(Hbf, mt * 16 + (lane & 15), FC, kk * KC, lane);
#pragma unroll
        for (int t = 0; t < 6; ++t) {
          v16bf bfrag = frag_ldB(Stage + (wave * 6 + t) * 512, lane);
          accO[mt][t] = __builtin_amdgcn_wmma_f32_16x16x32_bf16(
              false, afrag, false, bfrag, (short)0, accO[mt][t], false, false);
        }
      }
      __syncthreads();                    // readers done before next DMA
    }
  }

  // Epilogue: scatter-add weighted expert output.
  {
    const int half = (lane >> 4) * 8;
    const int nc   = lane & 15;
#pragma unroll
    for (int mt = 0; mt < 2; ++mt)
#pragma unroll
      for (int t = 0; t < 6; ++t) {
        int dcol = wave * 96 + t * 16 + nc;
        float b2v = b2[e * Dm + dcol];
#pragma unroll
        for (int r = 0; r < 8; ++r) {
          int row = mt * 16 + half + r;
          float w = rowW[row];
          if (w != 0.0f)
            atomicAdd(&out[(size_t)rowTok[row] * Dm + dcol],
                      w * (accO[mt][t][r] + b2v));
        }
      }
  }
}

extern "C" void kernel_launch(void* const* d_in, const int* in_sizes, int n_in,
                              void* d_out, int out_size, void* d_ws, size_t ws_size,
                              hipStream_t stream) {
  const float* x  = (const float*)d_in[0];
  const float* Wr = (const float*)d_in[1];
  const float* W1 = (const float*)d_in[2];
  const float* b1 = (const float*)d_in[3];
  const float* W2 = (const float*)d_in[4];
  const float* b2 = (const float*)d_in[5];
  float* out = (float*)d_out;

  char* ws = (char*)d_ws;
  int*            counts = (int*)(ws + WS_COUNTS);
  int*            tokIdx = (int*)(ws + WS_TIDX);
  float*          tokW   = (float*)(ws + WS_TW);
  unsigned short* Xbf    = (unsigned short*)(ws + WS_XBF);
  unsigned short* W1s    = (unsigned short*)(ws + WS_W1S);
  unsigned short* W2s    = (unsigned short*)(ws + WS_W2S);

  const int total = Ntok * Dm;
  moe_init_kernel<<<(total + 255) / 256, 256, 0, stream>>>(out, total, counts);

  const int prepN = 2 * Nexp * Dm * Fdim + Ntok * Dm;
  moe_prep_kernel<<<(prepN + 255) / 256, 256, 0, stream>>>(x, W1, W2,
                                                           Xbf, W1s, W2s);
  moe_router_kernel<<<Ntok / 8, 256, 0, stream>>>(x, Wr, counts, tokIdx, tokW);

  dim3 grid(Ntok / MT, Nexp);
  moe_expert_kernel<<<grid, 256, 0, stream>>>(Xbf, W1s, W2s, b1, b2,
                                              counts, tokIdx, tokW, out);
}
